// GAT_31817117729116
// MI455X (gfx1250) — matrix operand order repaired and checked by reading, not compiled, not run on hardware
//
#include <hip/hip_runtime.h>
#include <hip/hip_bf16.h>

typedef __attribute__((ext_vector_type(16))) _Float16 v16h;
typedef __attribute__((ext_vector_type(8)))  _Float16 h8;
typedef __attribute__((ext_vector_type(8)))  float    v8f;

#define NEG_SLOPE 0.2f

__device__ __forceinline__ void atomicMaxF(float* addr, float val) {
    int* ai = (int*)addr;
    int old = __float_as_int(*addr);
    while (__int_as_float(old) < val) {
        int assumed = old;
        old = atomicCAS(ai, assumed, __float_as_int(val));
        if (old == assumed) break;
    }
}

__device__ __forceinline__ float leaky(float l) {
    return (l > 0.f) ? l : NEG_SLOPE * l;
}

// ---------------- init per-call scratch (harness poisons ws with 0xAA) -------
__global__ void k_init(float* m1, float* s1, float* o1,
                       float* m2, float* s2, float* o2, int N) {
    int n = blockIdx.x * blockDim.x + threadIdx.x;
    if (n >= N) return;
    m1[2*n] = -3.0e38f; m1[2*n+1] = -3.0e38f;
    s1[2*n] = 0.f;      s1[2*n+1] = 0.f;
    m2[2*n] = -3.0e38f; m2[2*n+1] = -3.0e38f;
    s2[2*n] = 0.f;      s2[2*n+1] = 0.f;
#pragma unroll
    for (int j = 0; j < 32; ++j) o1[32*(size_t)n+j] = 0.f;
#pragma unroll
    for (int j = 0; j < 64; ++j) o2[64*(size_t)n+j] = 0.f;
}

// ---------------- conv1 node transform: h1 = x@W1 [N,32], att projections ----
__global__ void k_node1(const float* __restrict__ x, const float* __restrict__ W1,
                        const float* __restrict__ asw, const float* __restrict__ adw,
                        float* __restrict__ h1, float* __restrict__ as1,
                        float* __restrict__ ad1, int N) {
    __shared__ float sW[256], sAs[32], sAd[32];
    for (int i = threadIdx.x; i < 256; i += blockDim.x) sW[i] = W1[i];
    if (threadIdx.x < 32) { sAs[threadIdx.x] = asw[threadIdx.x]; sAd[threadIdx.x] = adw[threadIdx.x]; }
    __syncthreads();
    int n = blockIdx.x * blockDim.x + threadIdx.x;
    if (n >= N) return;
    float xv[8];
#pragma unroll
    for (int k = 0; k < 8; ++k) xv[k] = x[8*(size_t)n+k];
    float a_s[2] = {0.f, 0.f}, a_d[2] = {0.f, 0.f};
#pragma unroll
    for (int j = 0; j < 32; ++j) {
        float acc = 0.f;
#pragma unroll
        for (int k = 0; k < 8; ++k) acc += xv[k] * sW[k*32+j];
        h1[32*(size_t)n + j] = acc;
        a_s[j>>4] += acc * sAs[j];
        a_d[j>>4] += acc * sAd[j];
    }
    as1[2*n] = a_s[0]; as1[2*n+1] = a_s[1];
    ad1[2*n] = a_d[0]; ad1[2*n+1] = a_d[1];
}

// ---------------- conv2 node transform: h2 = o1@W2 [N,64] -------------------
__global__ void k_node2(const float* __restrict__ o1, const float* __restrict__ W2,
                        const float* __restrict__ asw, const float* __restrict__ adw,
                        float* __restrict__ h2, float* __restrict__ as2,
                        float* __restrict__ ad2, int N) {
    __shared__ float sW[2048], sAs[64], sAd[64];
    for (int i = threadIdx.x; i < 2048; i += blockDim.x) sW[i] = W2[i];
    for (int i = threadIdx.x; i < 64; i += blockDim.x) { sAs[i] = asw[i]; sAd[i] = adw[i]; }
    __syncthreads();
    int n = blockIdx.x * blockDim.x + threadIdx.x;
    if (n >= N) return;
    float xv[32];
#pragma unroll
    for (int k = 0; k < 32; ++k) xv[k] = o1[32*(size_t)n + k];
    float a_s[2] = {0.f, 0.f}, a_d[2] = {0.f, 0.f};
    for (int j = 0; j < 64; ++j) {
        float acc = 0.f;
#pragma unroll
        for (int k = 0; k < 32; ++k) acc += xv[k] * sW[k*64+j];
        h2[64*(size_t)n + j] = acc;
        a_s[j>>5] += acc * sAs[j];
        a_d[j>>5] += acc * sAd[j];
    }
    as2[2*n] = a_s[0]; as2[2*n+1] = a_s[1];
    ad2[2*n] = a_d[0]; ad2[2*n+1] = a_d[1];
}

// ---------------- softmax pass A: logits -> lbuf, per-dst max ----------------
__global__ void k_edge_max(const int* __restrict__ row, const int* __restrict__ col,
                           const float* __restrict__ as, const float* __restrict__ ad,
                           float* __restrict__ lbuf, float* __restrict__ mx,
                           int E, int N) {
    int i = blockIdx.x * blockDim.x + threadIdx.x;
    if (i >= E + N) return;
    int s = (i < E) ? row[i] : (i - E);
    int d = (i < E) ? col[i] : (i - E);
#pragma unroll
    for (int h = 0; h < 2; ++h) {
        float l = leaky(as[2*s+h] + ad[2*d+h]);
        lbuf[2*(size_t)i + h] = l;
        atomicMaxF(&mx[2*d+h], l);
    }
}

// ---------------- softmax pass B: lbuf -> exp (in place), per-dst sum --------
__global__ void k_edge_sum(const int* __restrict__ col,
                           const float* __restrict__ mx, float* __restrict__ lbuf,
                           float* __restrict__ sm, int E, int N) {
    int i = blockIdx.x * blockDim.x + threadIdx.x;
    if (i >= E + N) return;
    int d = (i < E) ? col[i] : (i - E);
#pragma unroll
    for (int h = 0; h < 2; ++h) {
        float ex = __expf(lbuf[2*(size_t)i + h] - mx[2*d+h]);
        lbuf[2*(size_t)i + h] = ex;
        atomicAdd(&sm[2*d+h], ex);
    }
}

// ---------------- softmax pass C: coalesced weighted scatter -----------------
// One wave32 per edge; channels across lanes: coalesced h[src] row read and
// lane-parallel, cacheline-contiguous atomic adds into o[dst].
__global__ void k_edge_agg32(const int* __restrict__ row, const int* __restrict__ col,
                             const float* __restrict__ lex, const float* __restrict__ sm,
                             const float* __restrict__ hsrc, float* __restrict__ oacc,
                             int E, int N) {
    long t = (long)blockIdx.x * blockDim.x + threadIdx.x;
    long i = t >> 5;
    int lane = (int)t & 31;
    if (i >= (long)E + N) return;
    int s = (i < E) ? row[i] : (int)(i - E);
    int d = (i < E) ? col[i] : (int)(i - E);
    int h = lane >> 4;                       // F=32: channel = lane, head = lane>>4
    float alpha = lex[2*i + h] / sm[2*d + h];
    atomicAdd(&oacc[(size_t)d*32 + lane], hsrc[(size_t)s*32 + lane] * alpha);
}

__global__ void k_edge_agg64(const int* __restrict__ row, const int* __restrict__ col,
                             const float* __restrict__ lex, const float* __restrict__ sm,
                             const float* __restrict__ hsrc, float* __restrict__ oacc,
                             int E, int N) {
    long t = (long)blockIdx.x * blockDim.x + threadIdx.x;
    long i = t >> 5;
    int lane = (int)t & 31;
    if (i >= (long)E + N) return;
    int s = (i < E) ? row[i] : (int)(i - E);
    int d = (i < E) ? col[i] : (int)(i - E);
    float a0 = lex[2*i + 0] / sm[2*d + 0];   // head 0: channels 0..31
    float a1 = lex[2*i + 1] / sm[2*d + 1];   // head 1: channels 32..63
    const float* hs = hsrc + (size_t)s*64;
    float* od = oacc + (size_t)d*64;
    atomicAdd(&od[lane],      hs[lane]      * a0);
    atomicAdd(&od[lane + 32], hs[lane + 32] * a1);
}

// ---------------- bias + relu (conv1 output, in place) -----------------------
__global__ void k_bias_relu(float* __restrict__ o, const float* __restrict__ b,
                            int total, int F) {
    int i = blockIdx.x * blockDim.x + threadIdx.x;
    if (i >= total) return;
    float v = o[i] + b[i % F];
    o[i] = v > 0.f ? v : 0.f;
}

// ---------------- conv2 finalize: bias+relu -> f16 ---------------------------
__global__ void k_finalize2(const float* __restrict__ o2, const float* __restrict__ b2,
                            _Float16* __restrict__ o2h, int total) {
    int i = blockIdx.x * blockDim.x + threadIdx.x;
    if (i >= total) return;
    float v = o2[i] + b2[i & 63];
    o2h[i] = (_Float16)(v > 0.f ? v : 0.f);
}

// ---------------- pack fc1_w [64,64] into WMMA B fragments (f16) -------------
// B 32x16 f16, contiguous k used here: lanes 0-15: N=lane, K=j; lanes 16-31: N=lane-16, K=16+j
__global__ void k_pack_b(const float* __restrict__ fc1w, _Float16* __restrict__ bfrag) {
    int t = threadIdx.x;                 // 256 threads: kt(2) x nt(4) x lane(32)
    int kt = t >> 7, nt = (t >> 5) & 3, lane = t & 31;
    int kl = (lane >= 16) ? 16 : 0;
    int nn = lane & 15;
#pragma unroll
    for (int j = 0; j < 16; ++j) {
        int k = kt * 32 + kl + j;
        int n = nt * 16 + nn;
        bfrag[(((kt*4 + nt) * 32) + lane) * 16 + j] = (_Float16)fc1w[k * 64 + n];
    }
}

// ---------------- edge MLP: z=h[r]+h[c]; relu(z@fc1+b1); @fc2+b2 ------------
// One wave32 per 16-edge tile; 8 x v_wmma_f32_16x16x32_f16 per tile.
__global__ void __launch_bounds__(256) k_edge_mlp(
        const int* __restrict__ row, const int* __restrict__ col,
        const _Float16* __restrict__ o2h, const _Float16* __restrict__ bfrag,
        const float* __restrict__ fc1b, const float* __restrict__ fc2w,
        const float* __restrict__ fc2b, float* __restrict__ out, int E) {
    __shared__ float lds[8][16][68];     // +4 pad, per-wave 16x64 tile
    __shared__ float sB1[64], sW2[64];
    if (threadIdx.x < 64) { sB1[threadIdx.x] = fc1b[threadIdx.x];
                            sW2[threadIdx.x] = fc2w[threadIdx.x]; }
    __syncthreads();

    int wave = threadIdx.x >> 5;
    int lane = threadIdx.x & 31;
    int g    = lane >> 4;                // lane half-group
    int m    = lane & 15;                // row within 16-edge tile
    long tile = (long)blockIdx.x * 8 + wave;
    long e    = tile * 16 + m;
    int r = 0, c = 0;
    if (e < E) { r = row[e]; c = col[e]; }
    const _Float16* pr = o2h + (size_t)r * 64;
    const _Float16* pc = o2h + (size_t)c * 64;

    // B fragments (weights, shared across edges)
    v16h bf[8];
#pragma unroll
    for (int kt = 0; kt < 2; ++kt)
#pragma unroll
        for (int nt = 0; nt < 4; ++nt)
            bf[kt*4+nt] = *(const v16h*)(bfrag + (((kt*4 + nt) * 32) + lane) * 16);

    v8f zero = {0.f,0.f,0.f,0.f,0.f,0.f,0.f,0.f};
    v8f acc[4] = {zero, zero, zero, zero};

#pragma unroll
    for (int kt = 0; kt < 2; ++kt) {
        // A 16x32 f16, contiguous k (same as B packing): lane group g holds K in [kt*32+g*16, +16)
        int base = kt * 32 + g * 16;
        h8 zr0 = *(const h8*)(pr + base);
        h8 zr1 = *(const h8*)(pr + base + 8);
        h8 zc0 = *(const h8*)(pc + base);
        h8 zc1 = *(const h8*)(pc + base + 8);
        h8 z0 = zr0 + zc0;
        h8 z1 = zr1 + zc1;
        v16h a;
#pragma unroll
        for (int j = 0; j < 8; ++j) { a[j] = z0[j]; a[8+j] = z1[j]; }
#pragma unroll
        for (int nt = 0; nt < 4; ++nt)
            acc[nt] = __builtin_amdgcn_wmma_f32_16x16x32_f16(
                false, a, false, bf[kt*4+nt], (short)0, acc[nt], false, false);
    }

    // D layout: VGPR v -> M = v + 8*g ; N = nt*16 + (lane&15). Bias+ReLU -> LDS
#pragma unroll
    for (int nt = 0; nt < 4; ++nt) {
        int n = nt * 16 + m;
        float bia = sB1[n];
#pragma unroll
        for (int v = 0; v < 8; ++v) {
            float val = acc[nt][v] + bia;
            lds[wave][g*8 + v][n] = val > 0.f ? val : 0.f;
        }
    }
    __syncthreads();

    // fc2: 64 -> 1 dot; lane does half the columns of its row, then xor-16 add
    float s = 0.f;
#pragma unroll
    for (int n = 0; n < 32; ++n) {
        int nn = g * 32 + n;
        s += lds[wave][m][nn] * sW2[nn];
    }
    s += __shfl_xor(s, 16, 32);
    if (g == 0 && e < E) out[e] = s + fc2b[0];
}

extern "C" void kernel_launch(void* const* d_in, const int* in_sizes, int n_in,
                              void* d_out, int out_size, void* d_ws, size_t ws_size,
                              hipStream_t stream) {
    (void)n_in; (void)out_size; (void)ws_size;
    const float* x    = (const float*)d_in[0];
    const int*   ei   = (const int*)d_in[1];
    const float* W1   = (const float*)d_in[2];
    const float* as1w = (const float*)d_in[3];
    const float* ad1w = (const float*)d_in[4];
    const float* b1   = (const float*)d_in[5];
    const float* W2   = (const float*)d_in[6];
    const float* as2w = (const float*)d_in[7];
    const float* ad2w = (const float*)d_in[8];
    const float* b2   = (const float*)d_in[9];
    const float* fc1w = (const float*)d_in[10];
    const float* fc1b = (const float*)d_in[11];
    const float* fc2w = (const float*)d_in[12];
    const float* fc2b = (const float*)d_in[13];

    int N = in_sizes[0] / 8;
    int E = in_sizes[1] / 2;
    const int* row = ei;          // edge_index[0]
    const int* col = ei + E;      // edge_index[1]

    // scratch carve-out (256B aligned)
    char* w = (char*)d_ws;
    auto alloc = [&](size_t bytes) -> char* {
        char* p = w; w += (bytes + 255) & ~(size_t)255; return p;
    };
    float*    h1   = (float*)alloc((size_t)N * 32 * 4);
    float*    as1  = (float*)alloc((size_t)N * 2 * 4);
    float*    ad1  = (float*)alloc((size_t)N * 2 * 4);
    float*    m1   = (float*)alloc((size_t)N * 2 * 4);
    float*    s1   = (float*)alloc((size_t)N * 2 * 4);
    float*    o1   = (float*)alloc((size_t)N * 32 * 4);
    float*    h2   = (float*)alloc((size_t)N * 64 * 4);
    float*    as2  = (float*)alloc((size_t)N * 2 * 4);
    float*    ad2  = (float*)alloc((size_t)N * 2 * 4);
    float*    m2   = (float*)alloc((size_t)N * 2 * 4);
    float*    s2   = (float*)alloc((size_t)N * 2 * 4);
    float*    o2   = (float*)alloc((size_t)N * 64 * 4);
    float*    lbuf = (float*)alloc(((size_t)E + N) * 2 * 4);  // logits -> exp, reused per conv
    _Float16* o2h  = (_Float16*)alloc((size_t)N * 64 * 2);
    _Float16* bfr  = (_Float16*)alloc(4096 * 2);

    const int tb = 256;
    int nbN = (N + tb - 1) / tb;
    int nbE = (E + N + tb - 1) / tb;
    int nbA = (int)(((long)(E + N) * 32 + tb - 1) / tb);   // wave-per-edge scatter

    k_init<<<nbN, tb, 0, stream>>>(m1, s1, o1, m2, s2, o2, N);

    // ---- conv1 ----
    k_node1<<<nbN, tb, 0, stream>>>(x, W1, as1w, ad1w, h1, as1, ad1, N);
    k_edge_max<<<nbE, tb, 0, stream>>>(row, col, as1, ad1, lbuf, m1, E, N);
    k_edge_sum<<<nbE, tb, 0, stream>>>(col, m1, lbuf, s1, E, N);
    k_edge_agg32<<<nbA, tb, 0, stream>>>(row, col, lbuf, s1, h1, o1, E, N);
    k_bias_relu<<<(N*32 + tb - 1) / tb, tb, 0, stream>>>(o1, b1, N*32, 32);

    // ---- conv2 ----
    k_node2<<<nbN, tb, 0, stream>>>(o1, W2, as2w, ad2w, h2, as2, ad2, N);
    k_edge_max<<<nbE, tb, 0, stream>>>(row, col, as2, ad2, lbuf, m2, E, N);
    k_edge_sum<<<nbE, tb, 0, stream>>>(col, m2, lbuf, s2, E, N);
    k_edge_agg64<<<nbA, tb, 0, stream>>>(row, col, lbuf, s2, h2, o2, E, N);
    k_finalize2<<<(N*64 + tb - 1) / tb, tb, 0, stream>>>(o2, b2, o2h, N*64);

    // ---- edge MLP head (WMMA) ----
    k_pack_b<<<1, 256, 0, stream>>>(fc1w, bfr);
    int tiles  = (E + 15) / 16;
    int blocks = (tiles + 7) / 8;
    k_edge_mlp<<<blocks, 256, 0, stream>>>(row, col, o2h, bfr, fc1b, fc2w, fc2b,
                                           (float*)d_out, E);
}
